// LSTMCell_49933289783734
// MI455X (gfx1250) — compile-verified, matrix operand order checked by reading, not run on hardware
//
#include <hip/hip_runtime.h>

#define BATCH  4096
#define HIDDEN 1024
#define KTOT   2048
#define FOURH  4096

#define BLK_M 128
#define BLK_N 32    // per gate
#define KT    32
#define NTILE (KTOT / KT)        // 64 K-steps total

// LDS row stride in halves (+8 pad; 80B rows keep 16B alignment, spread banks)
#define LDA 40
#define LDB 40
#define SA_SZ (BLK_M * LDA)          // 5120 halves
#define SB_SZ (4 * BLK_N * LDB)      // 5120 halves

typedef __attribute__((ext_vector_type(8)))  float  v8f;
typedef __attribute__((ext_vector_type(2)))  __bf16 v2bf;
typedef __attribute__((ext_vector_type(8)))  __bf16 v8bf;
typedef __attribute__((ext_vector_type(16))) __bf16 v16bf;

union BFU  { __bf16 b; unsigned short u; };
union BF2U { v2bf  v; unsigned int   u; };

static __device__ __forceinline__ unsigned int f2bf2(float lo, float hi) {
    BF2U t; t.v[0] = (__bf16)lo; t.v[1] = (__bf16)hi; return t.u;
}

static __device__ __forceinline__ float sigmoid_f(float x) {
    return 1.0f / (1.0f + __expf(-x));
}
static __device__ __forceinline__ float tanh_f(float x) {
    float e = __expf(2.0f * x);
    return (e - 1.0f) / (e + 1.0f);
}

static __device__ __forceinline__ v16bf ld_frag(const unsigned short* p0,
                                                const unsigned short* p1) {
    v8bf lo = *(const v8bf*)p0;
    v8bf hi = *(const v8bf*)p1;
    v16bf r;
#pragma unroll
    for (int i = 0; i < 8; ++i) { r[i] = lo[i]; r[i + 8] = hi[i]; }
    return r;
}

struct TileRegs { float4 a[4]; float4 b[4]; };

// Issue global b128 loads for one 32-deep K tile.
// A: 128x32 (8 fp32-quads per row); B: per-thread 4x4 micro-tile of one gate's
// 32K x 32N panel (4 consecutive K rows, coalesced float4 along N).
static __device__ __forceinline__ void load_tile(const float* __restrict__ Asrc,
                                                 const float* __restrict__ Bsrc,
                                                 int blockM, int blockN, int tid,
                                                 TileRegs& t) {
#pragma unroll
    for (int j = 0; j < 4; ++j) {
        int idx = tid + j * 256;          // 1024 float4 slots
        int row = idx >> 3;               // 0..127
        int kq  = (idx & 7) << 2;         // 0,4,..,28
        t.a[j] = *(const float4*)(Asrc + (size_t)(blockM + row) * HIDDEN + kq);
    }
    const int g  = tid >> 6;              // gate 0..3
    const int t6 = tid & 63;
    const int kq = (t6 >> 3) << 2;        // 0,4,..,28
    const int nq = (t6 & 7) << 2;         // 0,4,..,28
#pragma unroll
    for (int i = 0; i < 4; ++i)
        t.b[i] = *(const float4*)(Bsrc + (size_t)(kq + i) * FOURH
                                  + g * HIDDEN + blockN + nq);
}

// Convert to bf16 and stage into LDS.
// A: [M][K].  B: register-transposed 4x4 micro-tile -> [gate][N][K], so the
// transpose costs 4 ds_store_b64 instead of 16 ds_store_b16.
static __device__ __forceinline__ void store_tile(unsigned short* sA,
                                                  unsigned short* sB,
                                                  int tid, const TileRegs& t) {
#pragma unroll
    for (int j = 0; j < 4; ++j) {
        int idx = tid + j * 256;
        int row = idx >> 3;
        int kq  = (idx & 7) << 2;
        uint2 p;
        p.x = f2bf2(t.a[j].x, t.a[j].y);
        p.y = f2bf2(t.a[j].z, t.a[j].w);
        *(uint2*)(&sA[row * LDA + kq]) = p;
    }
    const int g  = tid >> 6;
    const int t6 = tid & 63;
    const int kq = (t6 >> 3) << 2;
    const int nq = (t6 & 7) << 2;
    unsigned short* base = &sB[g * (BLK_N * LDB)];
    const float* b0 = (const float*)&t.b[0];
    const float* b1 = (const float*)&t.b[1];
    const float* b2 = (const float*)&t.b[2];
    const float* b3 = (const float*)&t.b[3];
#pragma unroll
    for (int j = 0; j < 4; ++j) {          // n = nq + j
        uint2 p;
        p.x = f2bf2(b0[j], b1[j]);         // k = kq+0, kq+1
        p.y = f2bf2(b2[j], b3[j]);         // k = kq+2, kq+3
        *(uint2*)(&base[(nq + j) * LDB + kq]) = p;
    }
}

// 12 ds_load_b128 into fragments, then 8 WMMAs.
static __device__ __forceinline__ void compute_tile(const unsigned short* sA,
                                                    const unsigned short* sB,
                                                    int wm, int wn, int lane,
                                                    v8f acc[4][2]) {
    const int mlane = lane & 15;
    const int kl    = (lane >> 4) * 8;    // ISA 16-bit A-operand K split

    v16bf afrag[2];
#pragma unroll
    for (int mt = 0; mt < 2; ++mt) {
        int row = wm * 32 + mt * 16 + mlane;
        afrag[mt] = ld_frag(&sA[row * LDA + kl], &sA[row * LDA + 16 + kl]);
    }
    v16bf bfrag[4];
#pragma unroll
    for (int g = 0; g < 4; ++g) {
        int nrow = wn * 16 + mlane;
        const unsigned short* base = &sB[g * (BLK_N * LDB)];
        bfrag[g] = ld_frag(&base[nrow * LDB + kl], &base[nrow * LDB + 16 + kl]);
    }
#pragma unroll
    for (int g = 0; g < 4; ++g)
#pragma unroll
        for (int mt = 0; mt < 2; ++mt)
            acc[g][mt] = __builtin_amdgcn_wmma_f32_16x16x32_bf16(
                false, afrag[mt], false, bfrag[g], (short)0, acc[g][mt],
                false, false);
}

__global__ __launch_bounds__(256) void lstm_fused_wmma(
    const float* __restrict__ x, const float* __restrict__ h,
    const float* __restrict__ c, const float* __restrict__ W,
    const float* __restrict__ V, const float* __restrict__ b,
    float* __restrict__ out)
{
    // double-buffered staging: 2 x (10KB A + 10KB B) = 40KB of 320KB/WGP
    __shared__ unsigned short sA[2][SA_SZ];
    __shared__ unsigned short sB[2][SB_SZ];

    const int tid  = threadIdx.x;
    const int lane = tid & 31;
    const int wave = tid >> 5;
    const int wm   = wave >> 1;   // 0..3 -> 32 rows each
    const int wn   = wave & 1;    // 0..1 -> 16 cols each

    const int blockM = blockIdx.x * BLK_M;
    const int blockN = blockIdx.y * BLK_N;

    v8f acc[4][2];
#pragma unroll
    for (int g = 0; g < 4; ++g)
#pragma unroll
        for (int mt = 0; mt < 2; ++mt)
#pragma unroll
            for (int i = 0; i < 8; ++i) acc[g][mt][i] = 0.0f;

    // ---- software pipeline: preload tile 0 ----
    TileRegs t;
    load_tile(x, W, blockM, blockN, tid, t);
    store_tile(sA[0], sB[0], tid, t);
    __syncthreads();

    // Phase 0: x @ W (K tiles 0..31); Phase 1: h @ V (K tiles 32..63)
#pragma unroll 1
    for (int kt = 0; kt < NTILE; ++kt) {
        const int cur = kt & 1;
        if (kt + 1 < NTILE) {
            int k1 = (kt + 1) * KT;
            const float* Asrc = (k1 < HIDDEN) ? (x + k1) : (h + (k1 - HIDDEN));
            const float* Bsrc = (k1 < HIDDEN) ? (W + (size_t)k1 * FOURH)
                                              : (V + (size_t)(k1 - HIDDEN) * FOURH);
            load_tile(Asrc, Bsrc, blockM, blockN, tid, t);   // overlaps WMMAs below
        }

        compute_tile(sA[cur], sB[cur], wm, wn, lane, acc);

        if (kt + 1 < NTILE)
            store_tile(sA[cur ^ 1], sB[cur ^ 1], tid, t);
        __syncthreads();
    }

    // ---- epilogue: bias + gates + cell update, streaming (NT) traffic ----
    const int col = blockN + wn * 16 + (lane & 15);
    const float bi = b[0 * HIDDEN + col];
    const float bf = b[1 * HIDDEN + col];
    const float bo = b[2 * HIDDEN + col];
    const float bg = b[3 * HIDDEN + col];

    const int mhi = (lane >> 4) * 8;  // C/D layout: VGPR r -> M=r (+8 for lanes 16..31)

#pragma unroll
    for (int mt = 0; mt < 2; ++mt) {
#pragma unroll
        for (int r = 0; r < 8; ++r) {
            int row = blockM + wm * 32 + mt * 16 + r + mhi;
            float ig = sigmoid_f(acc[0][mt][r] + bi);
            float fg = sigmoid_f(acc[1][mt][r] + bf);
            float og = sigmoid_f(acc[2][mt][r] + bo);
            float gg = tanh_f   (acc[3][mt][r] + bg);

            size_t idx = (size_t)row * HIDDEN + col;
            float cin  = __builtin_nontemporal_load(&c[idx]);   // read-once
            float cnew = ig * gg + fg * cin;
            float hnew = og * tanh_f(cnew);

            __builtin_nontemporal_store(hnew, &out[idx]);                          // h_out
            __builtin_nontemporal_store(cnew, &out[(size_t)BATCH * HIDDEN + idx]); // c_out
        }
    }
}

extern "C" void kernel_launch(void* const* d_in, const int* in_sizes, int n_in,
                              void* d_out, int out_size, void* d_ws, size_t ws_size,
                              hipStream_t stream) {
    (void)in_sizes; (void)n_in; (void)out_size; (void)d_ws; (void)ws_size;
    const float* x = (const float*)d_in[0];
    const float* h = (const float*)d_in[1];
    const float* c = (const float*)d_in[2];
    const float* W = (const float*)d_in[3];
    const float* V = (const float*)d_in[4];
    const float* b = (const float*)d_in[5];
    float* out = (float*)d_out;

    dim3 grid(BATCH / BLK_M, HIDDEN / BLK_N);  // 32 x 32 blocks
    lstm_fused_wmma<<<grid, 256, 0, stream>>>(x, h, c, W, V, b, out);
}
